// HeteroGraph_65524021068291
// MI455X (gfx1250) — matrix operand order repaired and checked by reading, not compiled, not run on hardware
//
#include <hip/hip_runtime.h>
#include <hip/hip_bf16.h>

// ---------------- types for WMMA ----------------
typedef __attribute__((ext_vector_type(16))) __bf16 v16bf;
typedef __attribute__((ext_vector_type(8)))  __bf16 v8bf;
typedef __attribute__((ext_vector_type(8)))  float  v8f;

#define DIM 128
#define SMPITCH 136   // bf16 staging pitch (128 + 8 pad)
#define OUTPITCH 132  // f32 epilogue pitch (128 + 4 pad)

static __device__ __forceinline__ unsigned short f2bfbits(float x) {
    unsigned u = __float_as_uint(x);
    return (unsigned short)((u + 0x7FFFu + ((u >> 16) & 1u)) >> 16);  // RNE
}
static __device__ __forceinline__ __bf16 f2bf(float x) {
    unsigned short s = f2bfbits(x);
    __bf16 o;
    __builtin_memcpy(&o, &s, 2);
    return o;
}
static __device__ __forceinline__ unsigned packbf2(float lo, float hi) {
    return (unsigned)f2bfbits(lo) | ((unsigned)f2bfbits(hi) << 16);
}

// ---------------- utility kernels ----------------
__global__ void zero_kernel(float* __restrict__ p, size_t n) {
    size_t i = (size_t)blockIdx.x * blockDim.x + threadIdx.x;
    if (i < n) p[i] = 0.0f;
}

__global__ void degree_kernel(const int* __restrict__ src, const int* __restrict__ dst,
                              float* __restrict__ degOut, float* __restrict__ degIn, int E) {
    int e = blockIdx.x * blockDim.x + threadIdx.x;
    if (e >= E) return;
    atomicAdd(&degOut[src[e]], 1.0f);
    atomicAdd(&degIn[dst[e]], 1.0f);
}

__global__ void norm_kernel(float* __restrict__ degOut, float* __restrict__ degIn, int N) {
    int i = blockIdx.x * blockDim.x + threadIdx.x;
    if (i >= N) return;
    float a = degOut[i];
    degOut[i] = (a > 0.0f) ? rsqrtf(a) : 0.0f;
    float b = degIn[i];
    degIn[i] = (b > 0.0f) ? rsqrtf(b) : 0.0f;
}

// per-edge coefficient: c = nOut[src] * nIn[dst] (shared by both relations)
__global__ void coeff_kernel(const int* __restrict__ src, const int* __restrict__ dst,
                             const float* __restrict__ nOut, const float* __restrict__ nIn,
                             float* __restrict__ cbuf, int E) {
    int e = blockIdx.x * blockDim.x + threadIdx.x;
    if (e >= E) return;
    cbuf[e] = nOut[src[e]] * nIn[dst[e]];
}

// Pack W[l][r][k][n] (f32, row-major) into per-lane bf16 B-fragments for
// v_wmma_f32_16x16x32_bf16:  lane L holds column n = ntile*16 + (L&15),
// K = kstep*32 + (L>>4)*8 + e (e<8), kstep*32 + 16 + (L>>4)*8 + (e-8) (e>=8); 32B per lane.
__global__ void packW_kernel(const float* __restrict__ W, __bf16* __restrict__ Wpk) {
    int c = blockIdx.x * blockDim.x + threadIdx.x;     // chunk id
    const int NCHUNK = 2 * 3 * 8 * 4 * 32;
    if (c >= NCHUNK) return;
    int L  = c & 31;  int t0 = c >> 5;
    int ks = t0 & 3;  t0 >>= 2;
    int nt = t0 & 7;  t0 >>= 3;
    int r  = t0 % 3;  int l = t0 / 3;
    int hi = L >> 4, lm = L & 15;
    int n  = nt * 16 + lm;
    const float* wsrc = W + (size_t)(l * 3 + r) * DIM * DIM;
    __bf16* dst = Wpk + (size_t)c * 16;
    #pragma unroll
    for (int e = 0; e < 16; ++e) {
        int k = ks * 32 + (e < 8 ? hi * 8 + e : 16 + hi * 8 + (e - 8));
        dst[e] = f2bf(wsrc[(size_t)k * DIM + n]);
    }
}

// ---------------- fused 3-matrix WMMA GEMM ----------------
// Per block: 32 rows of h. 8 waves; wave w owns N-tile w (16 cols):
//   t0  = h@W0 + (b0+b1+b2)   (f32, coalesced float4 stores via LDS)
//   t1  = h@W1, t2 = h@W2     (packed bf16x2, coalesced uint4 stores via LDS)
__global__ __launch_bounds__(256) void gemm3_kernel(
    const float* __restrict__ hin, const __bf16* __restrict__ WpkL,
    const float* __restrict__ bvec,                // this layer's b: 3*128 f32
    float* __restrict__ t0, unsigned* __restrict__ t1pk, unsigned* __restrict__ t2pk,
    int N, int reluIn)
{
    __shared__ __align__(16) char smraw[32 * OUTPITCH * 4];   // 16896 B
    __bf16* sma = (__bf16*)smraw;   // staging: 32 x SMPITCH bf16 (8704 B)
    float*  smf = (float*)smraw;    // epilogue: 32 x OUTPITCH f32

    const int tid = threadIdx.x;
    const size_t rowBase = (size_t)blockIdx.x * 32;

    // cooperative load: 32 rows x 128 f32 -> bf16 LDS (optionally relu)
    #pragma unroll
    for (int j = 0; j < 4; ++j) {
        int idx = tid + j * 256;            // float4 index, 1024 total
        int row = idx >> 5;                 // 32 float4 per row
        int c4  = idx & 31;
        size_t grow = rowBase + row;
        float4 v = make_float4(0.f, 0.f, 0.f, 0.f);
        if (grow < (size_t)N)
            v = ((const float4*)(hin + grow * DIM))[c4];
        if (reluIn) {
            v.x = fmaxf(v.x, 0.f); v.y = fmaxf(v.y, 0.f);
            v.z = fmaxf(v.z, 0.f); v.w = fmaxf(v.w, 0.f);
        }
        __bf16* dp = &sma[row * SMPITCH + c4 * 4];
        dp[0] = f2bf(v.x); dp[1] = f2bf(v.y); dp[2] = f2bf(v.z); dp[3] = f2bf(v.w);
    }
    __syncthreads();

    const int lane = tid & 31;
    const int w    = tid >> 5;      // N-tile 0..7
    const int lm   = lane & 15;
    const int hi   = lane >> 4;

    v8f acc[2][3];
    #pragma unroll
    for (int mt = 0; mt < 2; ++mt)
        #pragma unroll
        for (int r = 0; r < 3; ++r)
            acc[mt][r] = (v8f)(0.0f);

    #pragma unroll
    for (int ks = 0; ks < 4; ++ks) {
        // A fragments from LDS (ISA 16-bit A layout)
        v16bf a[2];
        #pragma unroll
        for (int mt = 0; mt < 2; ++mt) {
            const __bf16* ap = &sma[(mt * 16 + lm) * SMPITCH + ks * 32 + hi * 8];
            v8bf lo = *(const v8bf*)ap;          // K = base .. base+7
            v8bf hv = *(const v8bf*)(ap + 16);   // K = base+16 .. base+23
            a[mt] = __builtin_shufflevector(lo, hv,
                0, 1, 2, 3, 4, 5, 6, 7, 8, 9, 10, 11, 12, 13, 14, 15);
        }
        // B fragments from packed W (contiguous 32B per lane, L2-resident)
        #pragma unroll
        for (int r = 0; r < 3; ++r) {
            const __bf16* bp = WpkL + ((((size_t)r * 8 + w) * 4 + ks) * 32 + lane) * 16;
            v16bf bfrag = *(const v16bf*)bp;
            acc[0][r] = __builtin_amdgcn_wmma_f32_16x16x32_bf16(
                false, a[0], false, bfrag, (short)0, acc[0][r], false, false);
            acc[1][r] = __builtin_amdgcn_wmma_f32_16x16x32_bf16(
                false, a[1], false, bfrag, (short)0, acc[1][r], false, false);
        }
    }

    const int n = w * 16 + lm;
    const float bsum = bvec[n] + bvec[DIM + n] + bvec[2 * DIM + n];

    __syncthreads();   // staging data dead; reuse LDS for epilogue

    // ---- round 0: t0 (f32) + bias, coalesced float4 stores ----
    #pragma unroll
    for (int mt = 0; mt < 2; ++mt)
        #pragma unroll
        for (int r = 0; r < 8; ++r)
            smf[(mt * 16 + hi * 8 + r) * OUTPITCH + n] = acc[mt][0][r] + bsum;
    __syncthreads();
    #pragma unroll
    for (int j = 0; j < 4; ++j) {
        int fi = tid + j * 256;
        int row = fi >> 5, c4 = fi & 31;
        size_t grow = rowBase + row;
        if (grow < (size_t)N) {
            float4 v = ((const float4*)(smf + row * OUTPITCH))[c4];
            ((float4*)(t0 + grow * DIM))[c4] = v;
        }
    }
    __syncthreads();

    // ---- rounds 1,2: t1/t2 packed bf16x2, coalesced uint4 stores ----
    #pragma unroll
    for (int m = 1; m <= 2; ++m) {
        unsigned* pkout = (m == 1) ? t1pk : t2pk;
        #pragma unroll
        for (int mt = 0; mt < 2; ++mt)
            #pragma unroll
            for (int r = 0; r < 8; ++r)
                smf[(mt * 16 + hi * 8 + r) * OUTPITCH + n] = acc[mt][m][r];
        __syncthreads();
        #pragma unroll
        for (int j = 0; j < 2; ++j) {
            int ui = tid + j * 256;          // uint4 index: 512 total
            int row = ui >> 4, q = ui & 15;  // 16 uint4 per row (8 cols each)
            size_t grow = rowBase + row;
            if (grow < (size_t)N) {
                const float* rp = smf + row * OUTPITCH + q * 8;
                uint4 o;
                o.x = packbf2(rp[0], rp[1]);
                o.y = packbf2(rp[2], rp[3]);
                o.z = packbf2(rp[4], rp[5]);
                o.w = packbf2(rp[6], rp[7]);
                ((uint4*)(pkout + grow * (DIM / 2)))[q] = o;
            }
        }
        __syncthreads();
    }
}

// ---------------- edge scatter: y[dst] += c*t1[src], y[src] += c*t2[dst] ----
// t1/t2 are packed bf16x2; each thread covers 2 dims of one edge.
__global__ void scatter_kernel(const int* __restrict__ src, const int* __restrict__ dst,
                               const unsigned* __restrict__ t1pk, const unsigned* __restrict__ t2pk,
                               const float* __restrict__ cbuf, float* __restrict__ y, int E) {
    size_t tid = (size_t)blockIdx.x * blockDim.x + threadIdx.x;
    int e = (int)(tid >> 6);
    int p = (int)(tid & 63);
    if (e >= E) return;
    float c = cbuf[e];
    if (c == 0.0f) return;
    int s = src[e], t = dst[e];
    unsigned u1 = t1pk[(size_t)s * (DIM / 2) + p];
    unsigned u2 = t2pk[(size_t)t * (DIM / 2) + p];
    float a0 = __uint_as_float(u1 << 16) * c;
    float a1 = __uint_as_float(u1 & 0xFFFF0000u) * c;
    float b0 = __uint_as_float(u2 << 16) * c;
    float b1 = __uint_as_float(u2 & 0xFFFF0000u) * c;
    float* yd = y + (size_t)t * DIM + p * 2;
    float* ys = y + (size_t)s * DIM + p * 2;
    atomicAdd(&yd[0], a0);
    atomicAdd(&yd[1], a1);
    atomicAdd(&ys[0], b0);
    atomicAdd(&ys[1], b1);
}

// ---------------- readout: relu + per-graph segment sum --------------------
__global__ void readout_kernel(const float* __restrict__ hfin, const int* __restrict__ gid,
                               float* __restrict__ outSum, float* __restrict__ cnt, int N) {
    size_t tid = (size_t)blockIdx.x * blockDim.x + threadIdx.x;
    int node = (int)(tid >> 7);
    int d = (int)(tid & 127);
    if (node >= N) return;
    int g = gid[node];
    float v = fmaxf(hfin[(size_t)node * DIM + d], 0.0f);
    atomicAdd(&outSum[(size_t)g * DIM + d], v);
    if (d == 0) atomicAdd(&cnt[g], 1.0f);
}

__global__ void divide_kernel(float* __restrict__ out, const float* __restrict__ cnt, int G) {
    size_t i = (size_t)blockIdx.x * blockDim.x + threadIdx.x;
    if (i >= (size_t)G * DIM) return;
    out[i] = out[i] / fmaxf(cnt[i >> 7], 1.0f);
}

// ---------------- host launcher --------------------------------------------
static inline size_t align256(size_t x) { return (x + 255) & ~(size_t)255; }

extern "C" void kernel_launch(void* const* d_in, const int* in_sizes, int n_in,
                              void* d_out, int out_size, void* d_ws, size_t ws_size,
                              hipStream_t stream) {
    const float* feat = (const float*)d_in[0];
    const float* W    = (const float*)d_in[1];
    const float* b    = (const float*)d_in[2];
    const int*   esrc = (const int*)d_in[3];
    const int*   edst = (const int*)d_in[4];
    const int*   gids = (const int*)d_in[5];

    const int N = in_sizes[0] / DIM;
    const int E = in_sizes[3];
    const int G = out_size / DIM;

    // workspace carve (~774 MB)
    char* p = (char*)d_ws;
    auto take = [&](size_t bytes) { char* q = p; p += align256(bytes); return q; };
    float*    degOut = (float*)take((size_t)N * 4);
    float*    degIn  = (float*)take((size_t)N * 4);
    float*    cbuf   = (float*)take((size_t)E * 4);
    float*    cnt    = (float*)take((size_t)G * 4);
    __bf16*   Wpk    = (__bf16*)take((size_t)2 * 3 * 8 * 4 * 32 * 16 * 2);
    unsigned* t1pk   = (unsigned*)take((size_t)N * (DIM / 2) * 4);
    unsigned* t2pk   = (unsigned*)take((size_t)N * (DIM / 2) * 4);
    float*    bufA   = (float*)take((size_t)N * DIM * 4);
    float*    bufB   = (float*)take((size_t)N * DIM * 4);
    float*    out    = (float*)d_out;

    const size_t WpkLayerStride = (size_t)3 * 8 * 4 * 32 * 16;  // bf16 elems

    // zero: degrees, counts, output accumulator
    {
        size_t n2 = (size_t)N;
        zero_kernel<<<(unsigned)((n2 + 255) / 256), 256, 0, stream>>>(degOut, n2);
        zero_kernel<<<(unsigned)((n2 + 255) / 256), 256, 0, stream>>>(degIn, n2);
        size_t og = (size_t)G * DIM;
        zero_kernel<<<(unsigned)((og + 255) / 256), 256, 0, stream>>>(out, og);
        zero_kernel<<<(unsigned)(((size_t)G + 255) / 256), 256, 0, stream>>>(cnt, (size_t)G);
    }

    degree_kernel<<<(E + 255) / 256, 256, 0, stream>>>(esrc, edst, degOut, degIn, E);
    norm_kernel<<<(N + 255) / 256, 256, 0, stream>>>(degOut, degIn, N);
    coeff_kernel<<<(E + 255) / 256, 256, 0, stream>>>(esrc, edst, degOut, degIn, cbuf, E);
    packW_kernel<<<(2 * 3 * 8 * 4 * 32 + 255) / 256, 256, 0, stream>>>(W, Wpk);

    const unsigned gemmGrid    = (unsigned)((N + 31) / 32);
    const unsigned scatterGrid = (unsigned)(((size_t)E * (DIM / 2) + 255) / 256);
    const unsigned nodeGrid    = (unsigned)(((size_t)N * DIM + 255) / 256);

    // ---- layer 0: input = feat (no relu), t0 -> bufA
    gemm3_kernel<<<gemmGrid, 256, 0, stream>>>(feat, Wpk, b,
                                               bufA, t1pk, t2pk, N, 0);
    scatter_kernel<<<scatterGrid, 256, 0, stream>>>(esrc, edst, t1pk, t2pk, cbuf, bufA, E);

    // ---- layer 1: input = relu(bufA), t0 -> bufB
    gemm3_kernel<<<gemmGrid, 256, 0, stream>>>(bufA, Wpk + WpkLayerStride, b + 3 * DIM,
                                               bufB, t1pk, t2pk, N, 1);
    scatter_kernel<<<scatterGrid, 256, 0, stream>>>(esrc, edst, t1pk, t2pk, cbuf, bufB, E);

    // ---- readout: relu(bufB) mean per graph
    readout_kernel<<<nodeGrid, 256, 0, stream>>>(bufB, gids, out, cnt, N);
    divide_kernel<<<(unsigned)(((size_t)G * DIM + 255) / 256), 256, 0, stream>>>(out, cnt, G);
}
